// MHA_63273458204822
// MI455X (gfx1250) — matrix operand order repaired and checked by reading, not compiled, not run on hardware
//
#include <hip/hip_runtime.h>

// MI455X (gfx1250) MHA forward: bf16 WMMA pipeline.
//   cvt(x,Wqkv,Wproj) -> QKV GEMM -> flash attention (S^T trick) -> proj GEMM
// All matmuls via v_wmma_f32_16x16x32_bf16, wave32.
// v3: GEMM inner loop reverted to round-1 form (B frag loaded per j) — that
//     version let the compiler unroll K x2 and software-pipeline loads under
//     WMMAs (72 static wmma, partial loadcnt waits). Attention keeps the v2
//     hoist of V^T loads above the softmax VALU block.

typedef __bf16 bf16_t;
typedef __attribute__((ext_vector_type(16))) __bf16 v16bf;
typedef __attribute__((ext_vector_type(8)))  __bf16 v8bf;
typedef __attribute__((ext_vector_type(4)))  __bf16 v4bf;
typedef __attribute__((ext_vector_type(8)))  float  v8f;

static constexpr int Bn = 4, T = 2048, D = 1024, H = 16, DK = 64;
static constexpr int M  = Bn * T;      // 8192 tokens
static constexpr int N_QKV = 3 * D;    // 3072

#define DEV __device__ __forceinline__

DEV v16bf cat8(v8bf lo, v8bf hi) {
  return __builtin_shufflevector(lo, hi, 0,1,2,3,4,5,6,7,8,9,10,11,12,13,14,15);
}

// A-matrix fragment (16x32 bf16, M x K), row-major source with leading dim ld.
// Lane layout: m = lane%16; element e <-> K = 16*(lane>=16) + e.
DEV v16bf load_a_frag(const bf16_t* __restrict__ base, int ld, int row0, int k0, int lane) {
  const int m  = row0 + (lane & 15);
  const int kA = k0 + ((lane >> 4) << 4);
  const bf16_t* p = base + (size_t)m * ld + kA;
  return cat8(*(const v8bf*)p, *(const v8bf*)(p + 8));
}

// B-matrix fragment (32x16 bf16, K x N) where memory holds B^T rows (N-major, K contiguous).
// Lane layout: n = lane%16; element e <-> K = 16*(lane>=16) + e.
DEV v16bf load_b_frag(const bf16_t* __restrict__ base, int ld, int col0, int k0, int lane) {
  const int n  = col0 + (lane & 15);
  const int kB = k0 + ((lane >> 4) << 4);
  const bf16_t* p = base + (size_t)n * ld + kB;
  return cat8(*(const v8bf*)p, *(const v8bf*)(p + 8));
}

DEV v8f wmma_bf16(v16bf a, v16bf b, v8f c) {
  return __builtin_amdgcn_wmma_f32_16x16x32_bf16(
      /*neg_a=*/false, a, /*neg_b=*/false, b,
      /*c_mod=*/(short)0, c, /*reuse_a=*/false, /*reuse_b=*/false);
}

// ---------------------------------------------------------------- f32 -> bf16
__global__ void k_cvt(const float* __restrict__ in, bf16_t* __restrict__ out, int n) {
  int i = (blockIdx.x * blockDim.x + threadIdx.x) * 4;
  if (i < n) {
    float4 f = *(const float4*)(in + i);
    v4bf v;
    v[0] = (bf16_t)f.x; v[1] = (bf16_t)f.y; v[2] = (bf16_t)f.z; v[3] = (bf16_t)f.w;
    *(v4bf*)(out + i) = v;
  }
}

// ------------------------------------------------------------------ QKV GEMM
// y[m, e] = sum_k x[m,k] * Wqkv[e,k];  scatter e into Q/K [b,h,t,d] and V^T [b,h,d,t].
__global__ __launch_bounds__(256) void k_gemm_qkv(
    const bf16_t* __restrict__ A, const bf16_t* __restrict__ W,
    bf16_t* __restrict__ Q, bf16_t* __restrict__ Kd, bf16_t* __restrict__ VT) {
  const int lane = threadIdx.x & 31;
  const int wid  = threadIdx.x >> 5;
  const int m0 = blockIdx.x * 64  + (wid >> 2) * 32;
  const int n0 = blockIdx.y * 256 + (wid & 3) * 64;
  v8f acc[2][4] = {};
  for (int k0 = 0; k0 < D; k0 += 32) {
    __builtin_prefetch(A + (size_t)(m0 + (lane & 15)) * D + k0 + 32, 0, 1);
    __builtin_prefetch(W + (size_t)(n0 + (lane & 15)) * D + k0 + 32, 0, 1);
    v16bf a0 = load_a_frag(A, D, m0,      k0, lane);
    v16bf a1 = load_a_frag(A, D, m0 + 16, k0, lane);
#pragma unroll
    for (int j = 0; j < 4; ++j) {
      v16bf bf = load_b_frag(W, D, n0 + j * 16, k0, lane);
      acc[0][j] = wmma_bf16(a0, bf, acc[0][j]);
      acc[1][j] = wmma_bf16(a1, bf, acc[1][j]);
    }
  }
  const int nq = lane & 15;
  const int mo = (lane >> 4) << 3;   // C layout: M = r + 8*(lane>=16)
#pragma unroll
  for (int i = 0; i < 2; ++i) {
#pragma unroll
    for (int j = 0; j < 4; ++j) {
      const int nn  = n0 + j * 16 + nq;
      const int sec = nn >> 10;            // 0=Q 1=K 2=V (uniform per subtile)
      const int rem = nn & (D - 1);
      const int h = rem >> 6, dd = rem & (DK - 1);
      const int mbase = m0 + i * 16 + mo;
      const int bb = mbase >> 11, tt = mbase & (T - 1);
      const size_t bh = (size_t)(bb * H + h);
      if (sec < 2) {
        bf16_t* dst = (sec == 0 ? Q : Kd) + (bh * T + tt) * DK + dd;
#pragma unroll
        for (int r = 0; r < 8; ++r) dst[(size_t)r * DK] = (bf16_t)acc[i][j][r];
      } else {
        v8bf pk;
#pragma unroll
        for (int r = 0; r < 8; ++r) pk[r] = (bf16_t)acc[i][j][r];
        *(v8bf*)(VT + (bh * DK + dd) * T + tt) = pk;   // t contiguous -> b128 store
      }
    }
  }
}

// -------------------------------------------------------------- attention
// One wave per (b,h,16-query tile). Computes S^T = K·Q^T so softmax over keys
// is per-lane (+ one xor-16 exchange); O^T = V^T · P^T keeps everything per-lane.
__global__ __launch_bounds__(256) void k_attn(
    const bf16_t* __restrict__ Q, const bf16_t* __restrict__ Kd,
    const bf16_t* __restrict__ VT, bf16_t* __restrict__ O) {
  const int lane = threadIdx.x & 31;
  const int w    = blockIdx.x * 8 + (threadIdx.x >> 5);
  const int bb = w >> 11;                 // H*(T/16) = 2048 waves per batch
  const int rem = w & 2047;
  const int h  = rem >> 7;                // T/16 = 128 q-tiles per head
  const int q0 = (rem & 127) * 16;
  const size_t off = (size_t)(bb * H + h) * (size_t)(T * DK);
  const bf16_t* Qh = Q  + off;
  const bf16_t* Kh = Kd + off;
  const bf16_t* Vh = VT + off;            // [DK][T]

  const v16bf qf0 = load_b_frag(Qh, DK, q0, 0,  lane);   // d = 0..31
  const v16bf qf1 = load_b_frag(Qh, DK, q0, 32, lane);   // d = 32..63

  v8f o[4] = {};
  float mrun = -1e30f, lrun = 0.f;
  const float scale = 0.125f;             // 1/sqrt(64)
  const bool hi = (lane >= 16);

  for (int j0 = 0; j0 < T; j0 += 32) {
    // S^T tiles: s0 = keys j0..j0+15, s1 = keys j0+16..j0+31 (M=key, N=query)
    v16bf ka0 = load_a_frag(Kh, DK, j0,      0,  lane);
    v16bf ka1 = load_a_frag(Kh, DK, j0,      32, lane);
    v16bf ka2 = load_a_frag(Kh, DK, j0 + 16, 0,  lane);
    v16bf ka3 = load_a_frag(Kh, DK, j0 + 16, 32, lane);
    v8f s0 = {}, s1 = {};
    s0 = wmma_bf16(ka0, qf0, s0);
    s0 = wmma_bf16(ka1, qf1, s0);
    s1 = wmma_bf16(ka2, qf0, s1);
    s1 = wmma_bf16(ka3, qf1, s1);

    // Hoist V^T fragment loads: independent of S, overlap with exp/shuffle VALU.
    v16bf va[4];
#pragma unroll
    for (int dt = 0; dt < 4; ++dt) va[dt] = load_a_frag(Vh, T, dt * 16, j0, lane);

    float mx = -1e30f;
#pragma unroll
    for (int r = 0; r < 8; ++r) {
      s0[r] *= scale; s1[r] *= scale;
      mx = fmaxf(mx, fmaxf(s0[r], s1[r]));
    }
    mx = fmaxf(mx, __shfl_xor(mx, 16, 32));
    const float mnew  = fmaxf(mrun, mx);
    const float alpha = __expf(mrun - mnew);
    lrun *= alpha;
#pragma unroll
    for (int dt = 0; dt < 4; ++dt)
#pragma unroll
      for (int r = 0; r < 8; ++r) o[dt][r] *= alpha;

    float p0[8], p1[8], ps = 0.f;
#pragma unroll
    for (int r = 0; r < 8; ++r) {
      p0[r] = __expf(s0[r] - mnew);
      p1[r] = __expf(s1[r] - mnew);
      ps += p0[r] + p1[r];
    }
    lrun += ps + __shfl_xor(ps, 16, 32);
    mrun = mnew;

    // Build P^T B-fragment: lane needs keys 16*(lane>=16)+e of this 32-block.
    v16bf pf;
#pragma unroll
    for (int r = 0; r < 8; ++r) {
      const float o0 = __shfl_xor(p0[r], 16, 32);
      const float o1 = __shfl_xor(p1[r], 16, 32);
      pf[r]     = (bf16_t)(hi ? o1 : p0[r]);   // keys base+0..7
      pf[r + 8] = (bf16_t)(hi ? p1[r] : o0);   // keys base+8..15
    }
#pragma unroll
    for (int dt = 0; dt < 4; ++dt) {
      o[dt] = wmma_bf16(va[dt], pf, o[dt]);
    }
  }

  const float inv = 1.0f / lrun;
  const int q = lane & 15;
  const int dbase = (lane >> 4) << 3;
  bf16_t* dst = O + (size_t)(bb * T + q0 + q) * D + h * DK + dbase;
#pragma unroll
  for (int dt = 0; dt < 4; ++dt) {
    v8bf pk;
#pragma unroll
    for (int r = 0; r < 8; ++r) pk[r] = (bf16_t)(o[dt][r] * inv);
    *(v8bf*)(dst + dt * 16) = pk;
  }
}

// ------------------------------------------------------------------ proj GEMM
__global__ __launch_bounds__(256) void k_gemm_proj(
    const bf16_t* __restrict__ A, const bf16_t* __restrict__ W,
    float* __restrict__ out) {
  const int lane = threadIdx.x & 31;
  const int wid  = threadIdx.x >> 5;
  const int m0 = blockIdx.x * 64  + (wid >> 2) * 32;
  const int n0 = blockIdx.y * 256 + (wid & 3) * 64;
  v8f acc[2][4] = {};
  for (int k0 = 0; k0 < D; k0 += 32) {
    __builtin_prefetch(A + (size_t)(m0 + (lane & 15)) * D + k0 + 32, 0, 1);
    __builtin_prefetch(W + (size_t)(n0 + (lane & 15)) * D + k0 + 32, 0, 1);
    v16bf a0 = load_a_frag(A, D, m0,      k0, lane);
    v16bf a1 = load_a_frag(A, D, m0 + 16, k0, lane);
#pragma unroll
    for (int j = 0; j < 4; ++j) {
      v16bf bf = load_b_frag(W, D, n0 + j * 16, k0, lane);
      acc[0][j] = wmma_bf16(a0, bf, acc[0][j]);
      acc[1][j] = wmma_bf16(a1, bf, acc[1][j]);
    }
  }
  const int nq = lane & 15;
  const int mo = (lane >> 4) << 3;
#pragma unroll
  for (int i = 0; i < 2; ++i) {
#pragma unroll
    for (int j = 0; j < 4; ++j) {
      float* dst = out + (size_t)(m0 + i * 16 + mo) * D + n0 + j * 16 + nq;
#pragma unroll
      for (int r = 0; r < 8; ++r) dst[(size_t)r * D] = acc[i][j][r];
    }
  }
}

// ------------------------------------------------------------------- launch
extern "C" void kernel_launch(void* const* d_in, const int* in_sizes, int n_in,
                              void* d_out, int out_size, void* d_ws, size_t ws_size,
                              hipStream_t stream) {
  const float* x     = (const float*)d_in[0];
  const float* Wqkv  = (const float*)d_in[1];
  const float* Wproj = (const float*)d_in[2];
  float* out = (float*)d_out;

  bf16_t* ws = (bf16_t*)d_ws;
  const size_t xN = (size_t)M * D;                    // 8M elems
  bf16_t* xb     = ws;  ws += xN;                     // 16 MB
  bf16_t* wqkvb  = ws;  ws += (size_t)N_QKV * D;      //  6 MB
  bf16_t* wprojb = ws;  ws += (size_t)D * D;          //  2 MB
  bf16_t* Qb     = ws;  ws += xN;                     // 16 MB
  bf16_t* Kb     = ws;  ws += xN;                     // 16 MB
  bf16_t* VTb    = ws;  ws += xN;                     // 16 MB
  bf16_t* Ob     = xb;                                // reuse x region after QKV GEMM

  k_cvt<<<(int)(xN / 1024), 256, 0, stream>>>(x, xb, (int)xN);
  k_cvt<<<(N_QKV * D) / 1024, 256, 0, stream>>>(Wqkv, wqkvb, N_QKV * D);
  k_cvt<<<(D * D) / 1024, 256, 0, stream>>>(Wproj, wprojb, D * D);

  k_gemm_qkv<<<dim3(M / 64, N_QKV / 256), 256, 0, stream>>>(xb, wqkvb, Qb, Kb, VTb);
  k_attn<<<(Bn * H * (T / 16)) / 8, 256, 0, stream>>>(Qb, Kb, VTb, Ob);
  k_gemm_proj<<<dim3(M / 64, D / 256), 256, 0, stream>>>(Ob, wprojb, out);
}
